// DecayMaskedMultiHeadAttention_17343077941561
// MI455X (gfx1250) — matrix operand order repaired and checked by reading, not compiled, not run on hardware
//
#include <hip/hip_runtime.h>
#include <hip/hip_bf16.h>

// ---------------------------------------------------------------------------
// DecayMaskedMultiHeadAttention for MI455X (gfx1250, wave32, WMMA)
//   B=4, N=1024, DIM=1024, H=16, Hd=64
// Pipeline:
//   1) x (f32) -> xh (f16)
//   2) wq/wk/wv/wo (f32) -> transposed f16 (Bt[k][n] = W[n][k])
//   3) WMMA GEMM: qh/kh/vh = xh @ W^T + b           (f16 out)
//   4) fused flash attention (decay mask pre-softmax) -> ctx (f16)
//   5) WMMA GEMM: out = ctx @ wo^T + bo             (f32 out)
// gfx1250 specifics: v_wmma_f32_16x16x32_f16 for all GEMMs,
// global_load_async_to_lds_b128 + s_wait_asynccnt for row-major tile staging,
// DPP ROW_ROR all-reduce for softmax row max/sum (no ds_bpermute).
// Workspace layout (<= 40 MB):
//   [0,8M)    xh, later reused as ctx
//   [8M,16M)  wqT/wkT/wvT/woT (2MB each)
//   [16M,40M) qh/kh/vh (8MB each)
// ---------------------------------------------------------------------------

typedef __attribute__((ext_vector_type(16))) _Float16 v16h;
typedef __attribute__((ext_vector_type(8)))  _Float16 v8h;
typedef __attribute__((ext_vector_type(8)))  float    v8f;

constexpr int DIMC = 1024;
constexpr int NTOK = 4096;   // B*N rows
constexpr int NSEQ = 1024;
constexpr int HD   = 64;

__device__ __forceinline__ v16h cat16(v8h lo, v8h hi) {
  return __builtin_shufflevector(lo, hi, 0,1,2,3,4,5,6,7,8,9,10,11,12,13,14,15);
}

// Async memory->LDS b128 copy (GV mode). For LDS destinations the low 32 bits
// of the generic pointer are the AS(3) offset (LLVM flat->local is trunc).
__device__ __forceinline__ void async_ld_b128(const void* gptr, void* lptr) {
  unsigned lds = (unsigned)(size_t)lptr;
  unsigned long long ga = (unsigned long long)(size_t)gptr;
  asm volatile("global_load_async_to_lds_b128 %0, %1, off"
               :: "v"(lds), "v"(ga) : "memory");
}
#define WAIT_ASYNC0() asm volatile("s_wait_asynccnt 0x0" ::: "memory")

// DPP ROW_ROR rotate within the 16-lane row (our C-layout "half").
#define DPP_ROR(x, n)                                                      \
  __builtin_bit_cast(float, __builtin_amdgcn_update_dpp(                   \
      0, __builtin_bit_cast(int, (x)), 0x120 + (n), 0xF, 0xF, true))

__device__ __forceinline__ float rowmax16(float x) {
  x = fmaxf(x, DPP_ROR(x, 1));
  x = fmaxf(x, DPP_ROR(x, 2));
  x = fmaxf(x, DPP_ROR(x, 4));
  x = fmaxf(x, DPP_ROR(x, 8));
  return x;
}
__device__ __forceinline__ float rowsum16(float x) {
  x += DPP_ROR(x, 1);
  x += DPP_ROR(x, 2);
  x += DPP_ROR(x, 4);
  x += DPP_ROR(x, 8);
  return x;
}

// ---------------------------------------------------------------- convert x
__global__ __launch_bounds__(256) void cvt_f16_kernel(
    const float* __restrict__ in, _Float16* __restrict__ out, int n) {
  int i = blockIdx.x * 256 + threadIdx.x;
  if (i < n) out[i] = (_Float16)in[i];
}

// ------------------------------------------------- transpose+convert weight
__global__ __launch_bounds__(256) void wtrans_kernel(
    const float* __restrict__ w, _Float16* __restrict__ wt) {
  __shared__ _Float16 tile[32][33];
  const int bx = blockIdx.x * 32, by = blockIdx.y * 32;
  const int tx = threadIdx.x & 31, ty = threadIdx.x >> 5;  // 8 rows / pass
#pragma unroll
  for (int i = 0; i < 32; i += 8)
    tile[ty + i][tx] = (_Float16)w[(size_t)(by + ty + i) * DIMC + bx + tx];
  __syncthreads();
#pragma unroll
  for (int i = 0; i < 32; i += 8)
    wt[(size_t)(bx + ty + i) * DIMC + by + tx] = tile[tx][ty + i];  // wt[k][n]=w[n][k]
}

// ------------------------------------------------------------ WMMA GEMM
// Y[M=4096][1024] = A[4096][1024] @ Bt[1024][1024] + bias   (Bt row-major K,N)
// block tile 128x64, 8 waves, wave tile 32x32 (2x2 WMMA), K step 32.
template <bool OUTF32>
__global__ __launch_bounds__(256) void gemm16_kernel(
    const _Float16* __restrict__ A, const _Float16* __restrict__ Bt,
    const float* __restrict__ bias, void* __restrict__ outp) {
  constexpr int LDA = 40;  // halves (80B rows, 16B aligned, conflict-spread)
  constexpr int LDB = 40;
  __shared__ _Float16 As[128 * LDA];       // [row m][k]
  __shared__ _Float16 Bs[64 * LDB];        // transposed: [n][k]

  const int tid   = threadIdx.x;
  const int lane  = tid & 31;
  const int w     = tid >> 5;
  const int wm    = w & 3;                 // M offset 32*wm
  const int wn    = w >> 2;                // N offset 32*wn
  const int m0    = blockIdx.y * 128;
  const int n0    = blockIdx.x * 64;
  const int lhalf = lane >> 4, lmod = lane & 15;

  v8f c[2][2];
#pragma unroll
  for (int i = 0; i < 2; ++i)
#pragma unroll
    for (int j = 0; j < 2; ++j)
#pragma unroll
      for (int v = 0; v < 8; ++v) c[i][j][v] = 0.f;

  for (int kk = 0; kk < DIMC; kk += 32) {
    __syncthreads();
    {  // A tile: 128 rows x 32 halves, async direct to LDS (2 x b128)
      const int r = tid >> 1, cq = (tid & 1) * 16;
      const _Float16* gp = A + (size_t)(m0 + r) * DIMC + kk + cq;
      async_ld_b128(gp,     &As[r * LDA + cq]);
      async_ld_b128(gp + 8, &As[r * LDA + cq + 8]);
    }
    {  // B tile: rows kk..kk+31 of Bt, cols n0..n0+63, stored transposed
      const int r = tid >> 3, cq = (tid & 7) * 8;
      const _Float16* gp = Bt + (size_t)(kk + r) * DIMC + n0 + cq;
      v8h x0 = *(const v8h*)gp;
#pragma unroll
      for (int i = 0; i < 8; ++i) Bs[(cq + i) * LDB + r] = x0[i];
    }
    WAIT_ASYNC0();
    __syncthreads();

    v16h af[2], bf[2];
#pragma unroll
    for (int i = 0; i < 2; ++i) {  // A frag: 16x32, row=lane&15
      const int row = wm * 32 + i * 16 + lmod;
      v8h lo = *(const v8h*)&As[row * LDA + lhalf * 8];
      v8h hi = *(const v8h*)&As[row * LDA + 16 + lhalf * 8];
      af[i] = cat16(lo, hi);
    }
#pragma unroll
    for (int j = 0; j < 2; ++j) {  // B frag: 32x16, K = 8*lhalf + h (h<8), 16 + 8*lhalf + h-8 (h>=8)
      const int nr = wn * 32 + j * 16 + lmod;
      v8h lo = *(const v8h*)&Bs[nr * LDB + lhalf * 8];
      v8h hi = *(const v8h*)&Bs[nr * LDB + 16 + lhalf * 8];
      bf[j] = cat16(lo, hi);
    }
#pragma unroll
    for (int i = 0; i < 2; ++i)
#pragma unroll
      for (int j = 0; j < 2; ++j)
        c[i][j] = __builtin_amdgcn_wmma_f32_16x16x32_f16(
            false, af[i], false, bf[j], (short)0, c[i][j], false, false);
  }

#pragma unroll
  for (int i = 0; i < 2; ++i)
#pragma unroll
    for (int j = 0; j < 2; ++j) {
      const int n = n0 + wn * 32 + j * 16 + lmod;
      const float bb = bias[n];
#pragma unroll
      for (int v = 0; v < 8; ++v) {
        const int m = m0 + wm * 32 + i * 16 + v + 8 * lhalf;
        const float val = c[i][j][v] + bb;
        if constexpr (OUTF32)
          ((float*)outp)[(size_t)m * DIMC + n] = val;
        else
          ((_Float16*)outp)[(size_t)m * DIMC + n] = (_Float16)val;
      }
    }
}

// ---------------------------------------------------- fused flash attention
// grid (N/64, H, B), 128 threads (4 waves x 16 query rows).
__global__ __launch_bounds__(128) void attn_kernel(
    const _Float16* __restrict__ Q, const _Float16* __restrict__ K,
    const _Float16* __restrict__ V, const float* __restrict__ dmask,
    _Float16* __restrict__ ctx) {
  constexpr int LQ = 72, LK = 72, LV = 40, LP = 40;   // padded LDS strides
  __shared__ _Float16 Qs[64 * LQ];      // [q row][hd]
  __shared__ _Float16 Ks[32 * LK];      // [key][hd]
  __shared__ _Float16 Vt[64 * LV];      // [hd][key]  (transposed)
  __shared__ _Float16 Ps[4][16 * LP];   // per-wave P staging [row][key]

  const int q0 = blockIdx.x * 64;
  const int h  = blockIdx.y;
  const int b  = blockIdx.z;
  const int tid = threadIdx.x, lane = tid & 31, w = tid >> 5;
  const int lhalf = lane >> 4, lmod = lane & 15;

  const size_t ho = (size_t)h * HD;
  const _Float16* Qg = Q + (size_t)b * NSEQ * DIMC + ho;
  const _Float16* Kg = K + (size_t)b * NSEQ * DIMC + ho;
  const _Float16* Vg = V + (size_t)b * NSEQ * DIMC + ho;
  const float*    Mg = dmask + (size_t)h * NSEQ * NSEQ;

  {  // load 64x64 Q block, async direct to LDS
    const int r = tid >> 1, cq = (tid & 1) * 32;
    const _Float16* gp = Qg + (size_t)(q0 + r) * DIMC + cq;
#pragma unroll
    for (int i = 0; i < 4; ++i)
      async_ld_b128(gp + 8 * i, &Qs[r * LQ + cq + 8 * i]);
  }
  WAIT_ASYNC0();
  __syncthreads();

  // Q A-frags (held all loop): hd chunks [0,32) and [32,64)
  v16h aq[2];
#pragma unroll
  for (int i = 0; i < 2; ++i) {
    const int row = w * 16 + lmod;
    v8h lo = *(const v8h*)&Qs[row * LQ + i * 32 + lhalf * 8];
    v8h hi = *(const v8h*)&Qs[row * LQ + i * 32 + 16 + lhalf * 8];
    aq[i] = cat16(lo, hi);
  }

  float Mrow[8], Lrow[8];
  v8f o[4];
#pragma unroll
  for (int v = 0; v < 8; ++v) { Mrow[v] = -1e30f; Lrow[v] = 0.f; }
#pragma unroll
  for (int j = 0; j < 4; ++j)
#pragma unroll
    for (int v = 0; v < 8; ++v) o[j][v] = 0.f;

  for (int kb = 0; kb < NSEQ; kb += 32) {
    __syncthreads();
    {  // K rows async to LDS; V via VGPRs, transposed into LDS
      const int r = tid >> 2, cq = (tid & 3) * 16;
      const _Float16* gk = Kg + (size_t)(kb + r) * DIMC + cq;
      async_ld_b128(gk,     &Ks[r * LK + cq]);
      async_ld_b128(gk + 8, &Ks[r * LK + cq + 8]);
      const _Float16* gv = Vg + (size_t)(kb + r) * DIMC + cq;
      v8h v0 = *(const v8h*)gv;
      v8h v1 = *(const v8h*)(gv + 8);
#pragma unroll
      for (int i = 0; i < 8; ++i) {
        Vt[(cq + i) * LV + r]     = v0[i];
        Vt[(cq + 8 + i) * LV + r] = v1[i];
      }
      if (kb + 32 < NSEQ) {  // speculative prefetch of next tiles
        __builtin_prefetch(Kg + (size_t)(kb + 32 + r) * DIMC + cq, 0, 1);
        __builtin_prefetch(Vg + (size_t)(kb + 32 + r) * DIMC + cq, 0, 1);
      }
    }
    WAIT_ASYNC0();
    __syncthreads();

    // ---- scores S = Q Kt  (two 16x16 tiles over 32 keys)
    v8f s[2];
#pragma unroll
    for (int t = 0; t < 2; ++t) {
      const int key = t * 16 + lmod;
      v8h lo = *(const v8h*)&Ks[key * LK + lhalf * 8];
      v8h hi = *(const v8h*)&Ks[key * LK + 16 + lhalf * 8];
      v16h bk0 = cat16(lo, hi);
      lo = *(const v8h*)&Ks[key * LK + 32 + lhalf * 8];
      hi = *(const v8h*)&Ks[key * LK + 32 + 16 + lhalf * 8];
      v16h bk1 = cat16(lo, hi);
      v8f z;
#pragma unroll
      for (int v = 0; v < 8; ++v) z[v] = 0.f;
      z = __builtin_amdgcn_wmma_f32_16x16x32_f16(false, aq[0], false, bk0,
                                                 (short)0, z, false, false);
      z = __builtin_amdgcn_wmma_f32_16x16x32_f16(false, aq[1], false, bk1,
                                                 (short)0, z, false, false);
      s[t] = z;
    }

    // ---- scale + decay mask (C layout: row = v + 8*lhalf, col = lmod+16t)
#pragma unroll
    for (int t = 0; t < 2; ++t)
#pragma unroll
      for (int v = 0; v < 8; ++v) {
        const int m = v + 8 * lhalf, n = t * 16 + lmod;
        const float dm = Mg[(size_t)(q0 + w * 16 + m) * NSEQ + kb + n];
        s[t][v] = s[t][v] * 0.125f * dm;
      }

    // ---- online softmax; row reductions via DPP ROW_ROR (VALU only)
    constexpr float L2E = 1.44269504088896f;
#pragma unroll
    for (int v = 0; v < 8; ++v) {
      const float x     = rowmax16(fmaxf(s[0][v], s[1][v]));
      const float newM  = fmaxf(Mrow[v], x);
      const float alpha = exp2f((Mrow[v] - newM) * L2E);
      Mrow[v] = newM;
      const float p0 = exp2f((s[0][v] - newM) * L2E);
      const float p1 = exp2f((s[1][v] - newM) * L2E);
      s[0][v] = p0; s[1][v] = p1;
      Lrow[v] = alpha * Lrow[v] + rowsum16(p0 + p1);
#pragma unroll
      for (int j = 0; j < 4; ++j) o[j][v] = o[j][v] * alpha;
    }

    // ---- P: C-layout -> A-layout via per-wave LDS stage (same-wave, in-order)
#pragma unroll
    for (int t = 0; t < 2; ++t)
#pragma unroll
      for (int v = 0; v < 8; ++v)
        Ps[w][(v + 8 * lhalf) * LP + t * 16 + lmod] = (_Float16)s[t][v];
    asm volatile("s_wait_dscnt 0" ::: "memory");
    v16h pa;
    {
      v8h lo = *(const v8h*)&Ps[w][lmod * LP + lhalf * 8];
      v8h hi = *(const v8h*)&Ps[w][lmod * LP + 16 + lhalf * 8];
      pa = cat16(lo, hi);
    }

    // ---- O += P @ V   (V frag from transposed LDS: b[h] = V[8*lhalf+h][n] (h<8), V[16+8*lhalf+h-8][n] (h>=8))
#pragma unroll
    for (int j = 0; j < 4; ++j) {
      const int hd = j * 16 + lmod;
      v8h lo = *(const v8h*)&Vt[hd * LV + lhalf * 8];
      v8h hi = *(const v8h*)&Vt[hd * LV + 16 + lhalf * 8];
      v16h bv = cat16(lo, hi);
      o[j] = __builtin_amdgcn_wmma_f32_16x16x32_f16(false, pa, false, bv,
                                                    (short)0, o[j], false, false);
    }
  }

  // ---- normalize + store context (f16)
#pragma unroll
  for (int j = 0; j < 4; ++j)
#pragma unroll
    for (int v = 0; v < 8; ++v) {
      const int m = q0 + w * 16 + v + 8 * lhalf;
      const int n = j * 16 + lmod;
      ctx[((size_t)b * NSEQ + m) * DIMC + ho + n] = (_Float16)(o[j][v] / Lrow[v]);
    }
}

// ---------------------------------------------------------------------------
extern "C" void kernel_launch(void* const* d_in, const int* in_sizes, int n_in,
                              void* d_out, int out_size, void* d_ws, size_t ws_size,
                              hipStream_t stream) {
  (void)in_sizes; (void)n_in; (void)out_size; (void)ws_size;
  const float* x   = (const float*)d_in[0];
  const float* dmk = (const float*)d_in[1];
  const float* wq  = (const float*)d_in[2];
  const float* bq  = (const float*)d_in[3];
  const float* wk  = (const float*)d_in[4];
  const float* bk  = (const float*)d_in[5];
  const float* wv  = (const float*)d_in[6];
  const float* bv  = (const float*)d_in[7];
  const float* wo  = (const float*)d_in[8];
  const float* bo  = (const float*)d_in[9];
  float* out = (float*)d_out;

  char* ws = (char*)d_ws;
  const size_t MB = 1u << 20;
  _Float16* xh  = (_Float16*)(ws);             // 8MB, reused as ctx later
  _Float16* wqT = (_Float16*)(ws + 8 * MB);
  _Float16* wkT = (_Float16*)(ws + 10 * MB);
  _Float16* wvT = (_Float16*)(ws + 12 * MB);
  _Float16* woT = (_Float16*)(ws + 14 * MB);
  _Float16* qh  = (_Float16*)(ws + 16 * MB);
  _Float16* kh  = (_Float16*)(ws + 24 * MB);
  _Float16* vh  = (_Float16*)(ws + 32 * MB);
  _Float16* ctx = xh;  // xh dead after QKV GEMMs

  const int nx = NTOK * DIMC;
  cvt_f16_kernel<<<nx / 256, 256, 0, stream>>>(x, xh, nx);

  dim3 tg(32, 32);
  wtrans_kernel<<<tg, 256, 0, stream>>>(wq, wqT);
  wtrans_kernel<<<tg, 256, 0, stream>>>(wk, wkT);
  wtrans_kernel<<<tg, 256, 0, stream>>>(wv, wvT);
  wtrans_kernel<<<tg, 256, 0, stream>>>(wo, woT);

  dim3 gg(DIMC / 64, NTOK / 128);  // (16, 32)
  gemm16_kernel<false><<<gg, 256, 0, stream>>>(xh, wqT, bq, qh);
  gemm16_kernel<false><<<gg, 256, 0, stream>>>(xh, wkT, bk, kh);
  gemm16_kernel<false><<<gg, 256, 0, stream>>>(xh, wvT, bv, vh);

  dim3 ag(NSEQ / 64, 16, 4);
  attn_kernel<<<ag, 128, 0, stream>>>(qh, kh, vh, dmk, ctx);

  gemm16_kernel<true><<<gg, 256, 0, stream>>>(ctx, woT, bo, (void*)out);
}
